// Decorrelation_35270271435435
// MI455X (gfx1250) — matrix operand order repaired and checked by reading, not compiled, not run on hardware
//
#include <hip/hip_runtime.h>
#include <hip/hip_bf16.h>

// ---------------------------------------------------------------------------
// Decorrelation kernel for MI455X (gfx1250).
// Memory-bound (AI ~0.004): 64MB lm stream dominates (floor ~3us @ 23.3TB/s).
// CDNA5 paths used:
//   - TDM tensor_load_to_lds + s_wait_tensorcnt: stage 27.8KB params + input
//     rows in LDS (kills ~450MB of per-block L2 re-read traffic)
//   - v_wmma_f32_16x16x32_f16: per-sample matvec correction out = x + L*x
//     (matrix pipe is idle in this roofline regime; f16 only on the ~1e-3
//     magnitude correction term, exact f32 for the identity part)
//   - nontemporal b128 stores for the write-once 64MB lm stream
// ---------------------------------------------------------------------------

typedef float f4 __attribute__((ext_vector_type(4)));
typedef float v8f __attribute__((ext_vector_type(8)));
typedef _Float16 v16h __attribute__((ext_vector_type(16)));
typedef unsigned int u32x4 __attribute__((ext_vector_type(4)));
typedef int i32x8 __attribute__((ext_vector_type(8)));
typedef int i32x4 __attribute__((ext_vector_type(4)));

#define NSAMP 8              // samples per block
#define NPAIR 496            // V*(V-1)/2, V=32
#define NBASIS 14
#define NPARAM (NBASIS * NPAIR)  // 6944

// Build a 1-D TDM descriptor (contiguous run of f32) and issue the DMA.
__device__ __forceinline__ void tdm_load_f32_1d(const float* gptr, float* lptr,
                                                unsigned nelem) {
  unsigned long long ga = (unsigned long long)(uintptr_t)gptr;
  unsigned lds = (unsigned)(uintptr_t)lptr;  // low 32 bits of flat = LDS byte addr

  u32x4 g0;
  g0[0] = 1u;                                    // count=1, is_restore=0, gather=0
  g0[1] = lds;                                   // lds_addr
  g0[2] = (unsigned)(ga & 0xFFFFFFFFu);          // global_addr[31:0]
  g0[3] = (unsigned)((ga >> 32) & 0x1FFFFFFu)    // global_addr[56:32]
        | (2u << 30);                            // type=2 ("image")

  i32x8 g1;
  g1[0] = (int)(2u << 16);                       // data_size=2 (4 bytes); mask=0
  g1[1] = (int)((nelem & 0xFFFFu) << 16);        // tensor_dim0[15:0]
  g1[2] = (int)((nelem >> 16) | (1u << 16));     // tensor_dim0[31:16], tensor_dim1=1
  g1[3] = (int)((nelem & 0xFFFFu) << 16);        // tile_dim0 = nelem (fits 16b)
  g1[4] = 1;                                     // tile_dim1=1, tile_dim2=0
  g1[5] = (int)nelem;                            // tensor_dim0_stride[31:0]
  g1[6] = 0;
  g1[7] = 0;

  i32x4 z4 = {0, 0, 0, 0};
#if __clang_major__ >= 23
  i32x8 z8 = {0, 0, 0, 0, 0, 0, 0, 0};
  __builtin_amdgcn_tensor_load_to_lds(g0, g1, z4, z4, z8, 0);
#else
  __builtin_amdgcn_tensor_load_to_lds(g0, g1, z4, z4, 0);
#endif
}

__global__ __launch_bounds__(512) void decor_main(
    const float* __restrict__ inp,    // (n, 32)
    const float* __restrict__ params, // (14, 496)
    float* __restrict__ out,          // (n, 32)
    float* __restrict__ lm,           // (n, 32, 32)
    int n) {
  __shared__ float sP[NPARAM];          // 27776 B
  __shared__ float sIn[NSAMP * 32];     //  1024 B
  __shared__ float sLam[NSAMP * NPAIR]; // 15872 B
  __shared__ float sCorr[NSAMP * 32];   //  1024 B  (L_strict @ x per sample)

  const int tid = threadIdx.x;
  const int n0 = blockIdx.x * NSAMP;
  if (n0 >= n) return;

  // ---- Phase 1: TDM async-stage params + 8 input rows into LDS ----
  if (tid == 0) {
    tdm_load_f32_1d(params, sP, NPARAM);
    tdm_load_f32_1d(inp + (size_t)n0 * 32, sIn, NSAMP * 32);
    __builtin_amdgcn_s_wait_tensorcnt((short)0);
  }
  __syncthreads();

  // ---- Phase 2: lam[s][p] via uniform cubic B-spline (closed form) ----
  if (tid < NPAIR) {
    const int p = tid;
    // recover (r, c): p = r*(r-1)/2 + c, 0 <= c < r
    int r = (int)(0.5f * (1.0f + sqrtf(8.0f * (float)p + 1.0f)));
    while (r * (r - 1) / 2 > p) --r;
    while ((r + 1) * r / 2 <= p) ++r;
    const int c = p - r * (r - 1) / 2;

    const float d = 30.0f / 11.0f;     // uniform knot spacing
    const float invd = 11.0f / 30.0f;
#pragma unroll
    for (int s = 0; s < NSAMP; ++s) {
      float x = sIn[s * 32 + c];
      x = fminf(fmaxf(x, -15.0f), 15.0f - 1e-6f);
      int j = 3 + (int)floorf((x + 15.0f) * invd);   // interval, j in [3,13]
      j = j < 3 ? 3 : (j > 13 ? 13 : j);
      float u = (x - (-15.0f + (float)(j - 3) * d)) * invd;  // local coord [0,1)
      float u2 = u * u, u3 = u2 * u;
      float om = 1.0f - u;
      float w0 = om * om * om * (1.0f / 6.0f);
      float w1 = (3.0f * u3 - 6.0f * u2 + 4.0f) * (1.0f / 6.0f);
      float w2 = (-3.0f * u3 + 3.0f * u2 + 3.0f * u + 1.0f) * (1.0f / 6.0f);
      float w3 = u3 * (1.0f / 6.0f);
      float lamv = w0 * sP[(j - 3) * NPAIR + p] + w1 * sP[(j - 2) * NPAIR + p] +
                   w2 * sP[(j - 1) * NPAIR + p] + w3 * sP[j * NPAIR + p];
      sLam[s * NPAIR + p] = lamv;
    }
  }
  __syncthreads();

  // ---- Phase 3a: WMMA matvec correction. Wave w -> sample s=w>>1, row-block
  //      rb=w&1. D(16x16,f32) = A(16x32,f16 strict-lower lm) x B(32x16,f16,
  //      col0 = x). One v_wmma_f32_16x16x32_f16 per wave, full EXEC. ----
  {
    const int wave = tid >> 5;
    const int lane = tid & 31;
    const int s = wave >> 1;
    const int rb = wave & 1;
    const int m = lane & 15;
    const bool hi = lane >= 16;
    const int v = rb * 16 + m;          // global row this lane feeds
    const int base = v * (v - 1) / 2;

    v16h A, B;
#pragma unroll
    for (int h = 0; h < 16; ++h) {
      // A layout: half h of lane l holds K = h + (h>=8?8:0) + (l>=16?8:0)
      int K = h + ((h >= 8) ? 8 : 0) + (hi ? 8 : 0);
      float a = (K < v) ? sLam[s * NPAIR + base + K] : 0.0f;
      A[h] = (_Float16)a;
      // B layout: same K as A: lane n half h = B[K][N=n], K = h + (h>=8?8:0) + (l>=16?8:0)
      float xv = (lane == 0)  ? sIn[s * 32 + h + ((h >= 8) ? 8 : 0)]
               : (lane == 16) ? sIn[s * 32 + 8 + h + ((h >= 8) ? 8 : 0)]
                              : 0.0f;
      B[h] = (_Float16)xv;
    }
    v8f C = {};
    C = __builtin_amdgcn_wmma_f32_16x16x32_f16(
        /*neg_a=*/false, A, /*neg_b=*/false, B,
        /*c_mod=*/(short)0, C, /*reuse_a=*/false, /*reuse_b=*/false);
    // Column N=0 of D: lane 0 holds M=0..7 (v0..v7), lane 16 holds M=8..15.
    if (lane == 0 || lane == 16) {
      int row0 = rb * 16 + (hi ? 8 : 0);
#pragma unroll
      for (int g = 0; g < 8; ++g) sCorr[s * 32 + row0 + g] = C[g];
    }
  }

  // ---- Phase 3b: stream lm (eye + strict-lower lam) with NT b128 stores ----
  for (int q = tid; q < NSAMP * 256; q += 512) {  // 256 float4 per 32x32 matrix
    int s = q >> 8;
    int e = q & 255;
    int v = e >> 3;
    int c0 = (e & 7) << 2;
    int base = v * (v - 1) / 2;
    f4 val;
#pragma unroll
    for (int mm = 0; mm < 4; ++mm) {
      int cc = c0 + mm;
      val[mm] = (cc == v) ? 1.0f
                          : (cc < v ? sLam[s * NPAIR + base + cc] : 0.0f);
    }
    f4* dst = (f4*)(lm + ((size_t)(n0 + s)) * 1024 + v * 32 + c0);
    __builtin_nontemporal_store(val, dst);
  }
  __syncthreads();

  // ---- Phase 4: out[n,v] = x[v] (exact f32) + correction (WMMA) ----
  if (tid < NSAMP * 32) {
    int s = tid >> 5;
    int v = tid & 31;
    out[(size_t)(n0 + s) * 32 + v] = sIn[s * 32 + v] + sCorr[s * 32 + v];
  }
}

// Penalty scalars from params only: one small block.
__global__ __launch_bounds__(256) void decor_pen(const float* __restrict__ params,
                                                 float* __restrict__ scal) {
  __shared__ float rs0[256], rs1[256], rs2[256];
  float a0 = 0.f, a1 = 0.f, a2 = 0.f;  // param_pen, first_pen, second_pen
  for (int p = threadIdx.x; p < NPAIR; p += 256) {
    float P[NBASIS];
#pragma unroll
    for (int k = 0; k < NBASIS; ++k) P[k] = params[k * NPAIR + p];
#pragma unroll
    for (int k = 0; k < NBASIS; ++k) a0 += P[k] * P[k];
#pragma unroll
    for (int k = 0; k < NBASIS - 1; ++k) {
      float dd = P[k + 1] - P[k];
      a1 += dd * dd;
    }
#pragma unroll
    for (int k = 0; k < NBASIS - 2; ++k) {
      float dd = P[k + 2] - 2.0f * P[k + 1] + P[k];
      a2 += dd * dd;
    }
  }
  const int t = threadIdx.x;
  rs0[t] = a0; rs1[t] = a1; rs2[t] = a2;
  __syncthreads();
  for (int off = 128; off > 0; off >>= 1) {
    if (t < off) {
      rs0[t] += rs0[t + off];
      rs1[t] += rs1[t + off];
      rs2[t] += rs2[t + off];
    }
    __syncthreads();
  }
  if (t == 0) {
    scal[0] = rs2[0];  // second_pen
    scal[1] = rs1[0];  // first_pen
    scal[2] = rs0[0];  // param_pen
  }
}

extern "C" void kernel_launch(void* const* d_in, const int* in_sizes, int n_in,
                              void* d_out, int out_size, void* d_ws, size_t ws_size,
                              hipStream_t stream) {
  const float* inp = (const float*)d_in[0];     // (n, 32) f32
  // d_in[1] = log_d: unused by the reference computation
  const float* params = (const float*)d_in[2];  // (14, 496) f32

  const int n = in_sizes[0] / 32;               // 16384
  float* out = (float*)d_out;                   // (n, 32)
  float* lm = out + (size_t)n * 32;             // (n, 32, 32)
  float* scal = lm + (size_t)n * 32 * 32;       // 3 scalars

  decor_main<<<n / NSAMP, 512, 0, stream>>>(inp, params, out, lm, n);
  decor_pen<<<1, 256, 0, stream>>>(params, scal);
}
